// SelfAttention_27977416966418
// MI455X (gfx1250) — matrix operand order repaired and checked by reading, not compiled, not run on hardware
//
#include <hip/hip_runtime.h>
#include <hip/hip_bf16.h>
#include <math.h>

// ---------------------------------------------------------------------------
// Self-attention (B=2, T=2048, D=2048, H=16, HD=128) for gfx1250 (MI455X).
// All GEMMs run through V_WMMA_F32_16X16X32_BF16 (bf16 in, f32 accumulate).
// Attention K tiles stream global->LDS via the Tensor Data Mover (prefetched
// one tile ahead, TENSORcnt-synchronized); GEMM tiles use a double-buffered
// LDS pipeline with register staging (one barrier per K step).
// ---------------------------------------------------------------------------

typedef __bf16 bf16;
typedef __attribute__((ext_vector_type(16))) __bf16 v16bf;
typedef __attribute__((ext_vector_type(8)))  float  v8f;
typedef __attribute__((ext_vector_type(4)))  unsigned int u32x4;
typedef __attribute__((ext_vector_type(4)))  int i32x4;
typedef __attribute__((ext_vector_type(8)))  int i32x8;

#define BB  2
#define TT  2048
#define DD  2048
#define HH  16
#define HD  128
#define D3  6144
#define BT  (BB * TT)        // 4096 rows

#if defined(__has_builtin)
#if __has_builtin(__builtin_amdgcn_tensor_load_to_lds) && \
    __has_builtin(__builtin_amdgcn_s_wait_tensorcnt)
#define HAVE_TDM 1
#endif
#endif

// ---------------------------------------------------------------------------
// TDM: issue a 2-D tile load global->LDS, 2-byte elements (no wait).
// D# group0: [1:0]=count=1, [63:32]=lds_addr, [120:64]=global_addr,
//            [127:126]=type=2.
// D# group1: [17:16]=data_size=1(2B), [79:48]=tensor_dim0, [111:80]=
//            tensor_dim1, [127:112]=tile_dim0, [143:128]=tile_dim1,
//            [207:160]=tensor_dim0_stride.
// ---------------------------------------------------------------------------
__device__ __forceinline__ void tdm_issue_2d(unsigned lds_off,
                                             const void* gptr,
                                             unsigned tile_w,
                                             unsigned tile_h,
                                             unsigned stride_elems) {
#ifdef HAVE_TDM
  unsigned long long ga = (unsigned long long)gptr;
  u32x4 g0;
  g0[0] = 1u;                                    // count = 1 (valid user D#)
  g0[1] = lds_off;                               // lds_addr (bytes)
  g0[2] = (unsigned)ga;                          // global_addr[31:0]
  g0[3] = (unsigned)((ga >> 32) & 0x01ffffffu)   // global_addr[56:32]
          | 0x80000000u;                         // type = 2 ("image")
  i32x8 g1;
  g1[0] = (int)(1u << 16);                       // data_size = 1 -> 2 bytes
  g1[1] = (int)((tile_w & 0xffffu) << 16);       // tensor_dim0[15:0]
  g1[2] = (int)(((tile_w >> 16) & 0xffffu) |     // tensor_dim0[31:16]
                ((tile_h & 0xffffu) << 16));     // tensor_dim1[15:0]
  g1[3] = (int)(((tile_h >> 16) & 0xffffu) |     // tensor_dim1[31:16]
                ((tile_w & 0xffffu) << 16));     // tile_dim0
  g1[4] = (int)(tile_h & 0xffffu);               // tile_dim1 (tile_dim2 = 0)
  g1[5] = (int)stride_elems;                     // tensor_dim0_stride[31:0]
  g1[6] = 0;                                     // stride hi / dim1 stride lo
  g1[7] = 0;
  i32x4 gz = {0, 0, 0, 0};
#if __clang_major__ >= 23
  i32x8 gz8 = {0, 0, 0, 0, 0, 0, 0, 0};
  __builtin_amdgcn_tensor_load_to_lds(g0, g1, gz, gz, gz8, 0);
#else
  __builtin_amdgcn_tensor_load_to_lds(g0, g1, gz, gz, 0);
#endif
#else
  (void)lds_off; (void)gptr; (void)tile_w; (void)tile_h; (void)stride_elems;
#endif
}

// ---------------------------------------------------------------------------
// WMMA fragment loaders (CDNA5 ISA 7.12.2 layouts), usable on LDS or global.
// A (16x32, MxK):  lane&15 = M; lanes 0-15 hold K 0-7 & 16-23, lanes 16-31
//                  hold K 8-15 & 24-31.
// B (32x16, KxN):  lane&15 = N; lanes 0-15 hold K 0-7 & 16-23, lanes 16-31
//                  hold K 8-15 & 24-31 (same as A). Source indexed as [n][k].
// C/D (16x16):     VGPR r, lane l -> M = r + 8*(l>>4), N = l&15.
// ---------------------------------------------------------------------------
__device__ __forceinline__ v16bf load_a_frag(const bf16* p, int stride,
                                             int m_off, int k_off) {
  const int lane = threadIdx.x & 31;
  const int half = lane >> 4;
  const int m    = lane & 15;
  const bf16* row = p + (size_t)(m_off + m) * stride + k_off;
  v16bf r;
#pragma unroll
  for (int v = 0; v < 8; ++v) {
    int k = ((v & 3) << 1) + (half << 3) + ((v >> 2) << 4);
    r[2 * v]     = row[k];
    r[2 * v + 1] = row[k + 1];
  }
  return r;
}

__device__ __forceinline__ v16bf load_b_frag(const bf16* p, int stride,
                                             int n_off, int k_off) {
  const int lane  = threadIdx.x & 31;
  const int khalf = lane >> 4;
  const int n     = lane & 15;
  const bf16* row = p + (size_t)(n_off + n) * stride + k_off;
  v16bf r;
#pragma unroll
  for (int v = 0; v < 8; ++v) {
    int k = ((v & 3) << 1) + (khalf << 3) + ((v >> 2) << 4);
    r[2 * v]     = row[k];
    r[2 * v + 1] = row[k + 1];
  }
  return r;
}

__device__ __forceinline__ v8f wmma_bf16(v16bf a, v16bf b, v8f c) {
  return __builtin_amdgcn_wmma_f32_16x16x32_bf16(false, a, false, b,
                                                 (short)0, c, false, false);
}

__device__ __forceinline__ void cvt4(bf16* dst, float4 f) {
  bf16 h4[4] = {(bf16)f.x, (bf16)f.y, (bf16)f.z, (bf16)f.w};
  *(uint2*)dst = *(uint2*)h4;
}

// ---------------------------------------------------------------------------
// Kernel 1: RoPE cos/sin tables, f32 [T][64]
// ---------------------------------------------------------------------------
__global__ void rope_table_kernel(float* __restrict__ ct,
                                  float* __restrict__ st) {
  const int t = blockIdx.x;
  const int d = threadIdx.x;          // 0..63
  float inv_freq = powf(10000.0f, -(float)d / 64.0f);
  float f = (float)t * inv_freq;
  ct[t * 64 + d] = cosf(f);
  st[t * 64 + d] = sinf(f);
}

// ---------------------------------------------------------------------------
// Kernel 2: qkv = x @ Wqkv^T + bqkv, scattered into head-major Q/K/V (bf16).
// Block tile 128(M) x 64(N), 8 waves; double-buffered LDS, one barrier/step.
// ---------------------------------------------------------------------------
__global__ __launch_bounds__(256) void gemm_qkv_kernel(
    const float* __restrict__ x, const float* __restrict__ w,
    const float* __restrict__ bias, bf16* __restrict__ qb,
    bf16* __restrict__ kb, bf16* __restrict__ vb) {
  __shared__ bf16 As[2][128 * 32];
  __shared__ bf16 Bs[2][64 * 32];
  const int tid  = threadIdx.x;
  const int wave = tid >> 5;
  const int lane = tid & 31;
  const int half = lane >> 4;
  const int m0 = blockIdx.x * 128;
  const int n0 = blockIdx.y * 64;

  const v8f z8 = {0.f, 0.f, 0.f, 0.f, 0.f, 0.f, 0.f, 0.f};
  v8f acc[4] = {z8, z8, z8, z8};

  float4 fa[4], fb[2];
  auto load_tiles = [&](int k0) {
#pragma unroll
    for (int i = 0; i < 4; ++i) {
      int e = (tid + i * 256) * 4;
      int r = e >> 5, c = e & 31;
      fa[i] = *(const float4*)(x + (size_t)(m0 + r) * DD + k0 + c);
    }
#pragma unroll
    for (int i = 0; i < 2; ++i) {
      int e = (tid + i * 256) * 4;
      int r = e >> 5, c = e & 31;
      fb[i] = *(const float4*)(w + (size_t)(n0 + r) * DD + k0 + c);
    }
  };
  auto store_tiles = [&](int buf) {
#pragma unroll
    for (int i = 0; i < 4; ++i) {
      int e = (tid + i * 256) * 4;
      cvt4(&As[buf][e], fa[i]);
    }
#pragma unroll
    for (int i = 0; i < 2; ++i) {
      int e = (tid + i * 256) * 4;
      cvt4(&Bs[buf][e], fb[i]);
    }
  };

  load_tiles(0);
  store_tiles(0);
  int buf = 0;
  for (int k0 = 0; k0 < DD; k0 += 32) {
    __syncthreads();                       // publish tile k0 in As/Bs[buf]
    const bool more = (k0 + 32 < DD);
    if (more) load_tiles(k0 + 32);         // global loads overlap WMMA below
    v16bf a = load_a_frag(As[buf], 32, wave * 16, 0);
#pragma unroll
    for (int j = 0; j < 4; ++j) {
      v16bf b = load_b_frag(Bs[buf], 32, j * 16, 0);
      acc[j] = wmma_bf16(a, b, acc[j]);
    }
    if (more) { buf ^= 1; store_tiles(buf); }
  }

  // Epilogue. A 64-wide N tile never crosses a q/k/v section (2048 % 64 == 0)
  // nor a head boundary (64 | 128), so which/h/d-base are block-uniform.
  const int which = n0 >> 11;                    // 0=q 1=k 2=v
  const int h     = (n0 & 2047) >> 7;
  const int dbase = n0 & 127;
  bf16* dst = (which == 0) ? qb : ((which == 1) ? kb : vb);
#pragma unroll
  for (int j = 0; j < 4; ++j) {
#pragma unroll
    for (int r = 0; r < 8; ++r) {
      int m = m0 + wave * 16 + r + 8 * half;     // global row = b*T + t
      int n = n0 + j * 16 + (lane & 15);
      float val = acc[j][r] + bias[n];
      int b = m >> 11, t = m & 2047;
      int d = dbase + j * 16 + (lane & 15);
      dst[((size_t)((b * HH + h) * TT + t) << 7) + d] = (bf16)val;
    }
  }
}

// ---------------------------------------------------------------------------
// Kernel 3: apply rotary embedding in-place to Q and K (bf16, f32 math).
// 256 threads = 4 rows of 64 rotation pairs.
// ---------------------------------------------------------------------------
__global__ __launch_bounds__(256) void rope_apply_kernel(
    bf16* __restrict__ qb, bf16* __restrict__ kb,
    const float* __restrict__ ct, const float* __restrict__ st) {
  const int idx = blockIdx.x * 256 + threadIdx.x;  // over B*H*T*64
  const int row = idx >> 6;                        // bh*T + t
  const int t   = row & (TT - 1);
  const int d   = idx & 63;
  const size_t base = (size_t)row * HD;
  float c = ct[t * 64 + d], s = st[t * 64 + d];

  float q1 = (float)qb[base + d], q2 = (float)qb[base + d + 64];
  qb[base + d]      = (bf16)(q1 * c - q2 * s);
  qb[base + d + 64] = (bf16)(q1 * s + q2 * c);

  float k1 = (float)kb[base + d], k2 = (float)kb[base + d + 64];
  kb[base + d]      = (bf16)(k1 * c - k2 * s);
  kb[base + d + 64] = (bf16)(k1 * s + k2 * c);
}

// ---------------------------------------------------------------------------
// Kernel 4: causal flash attention. Block = (b,h) x 128 query rows, 8 waves,
// each wave one 16-row strip. Q fragments live in registers; K tiles stream
// via the TDM into ping-pong LDS buffers, prefetched one tile ahead while
// all waves run WMMA; V staged transposed; P re-enters WMMA A-layout via LDS.
// ---------------------------------------------------------------------------
__global__ __launch_bounds__(256) void attn_kernel(
    const bf16* __restrict__ qb, const bf16* __restrict__ kb,
    const bf16* __restrict__ vb, bf16* __restrict__ ob) {
  __shared__ bf16 Ks[2][64 * 128]; // [keypos][d] ping-pong
  __shared__ bf16 Vt[128 * 64];    // [d][keypos]
  __shared__ bf16 Ps[8 * 16 * 64]; // per-wave P staging [m][n]

  const int tid  = threadIdx.x;
  const int wave = tid >> 5;
  const int lane = tid & 31;
  const int half = lane >> 4;
  const int q0 = blockIdx.x * 128;
  const int bh = blockIdx.y;                 // b*H + h
  const int b = bh >> 4, h = bh & 15;

  const bf16* qrow = qb + (size_t)bh * TT * HD;
  const bf16* krow = kb + (size_t)bh * TT * HD;
  const bf16* vrow = vb + (size_t)bh * TT * HD;

  // Preload this wave's 16x128 Q strip as 4 A-fragments (registers).
  v16bf qf[4];
  {
    const int m = lane & 15;
    const bf16* qp = qrow + (size_t)(q0 + wave * 16 + m) * HD;
#pragma unroll
    for (int kk = 0; kk < 4; ++kk) {
#pragma unroll
      for (int v = 0; v < 8; ++v) {
        int k = kk * 32 + ((v & 3) << 1) + (half << 3) + ((v >> 2) << 4);
        qf[kk][2 * v]     = qp[k];
        qf[kk][2 * v + 1] = qp[k + 1];
      }
    }
  }

  const v8f z8 = {0.f, 0.f, 0.f, 0.f, 0.f, 0.f, 0.f, 0.f};
  v8f o[8] = {z8, z8, z8, z8, z8, z8, z8, z8};
  float mrow[8], lrow[8];
#pragma unroll
  for (int r = 0; r < 8; ++r) { mrow[r] = -1e30f; lrow[r] = 0.f; }
  const float scale = 0.08838834764831845f;  // 1/sqrt(128)

  const int ktend = q0 + 128;
#ifdef HAVE_TDM
  if (wave == 0) {   // TDM prologue: start streaming K tile 0
    tdm_issue_2d((unsigned)(size_t)&Ks[0][0], krow, HD, 64, HD);
  }
#endif
  int buf = 0;
  for (int kt = 0; kt < ktend; kt += 64) {
#ifndef HAVE_TDM
    for (int i = tid; i < (64 * 128) / 8; i += 256) {
      *(uint4*)(Ks[buf] + i * 8) =
          *(const uint4*)(krow + (size_t)kt * HD + i * 8);
    }
#endif
    // V transposed: batch 8 uint2 global loads, then strided LDS stores.
    {
      uint2 vreg[8];
#pragma unroll
      for (int i = 0; i < 8; ++i) {
        int idx = tid + i * 256;
        int t = idx >> 5, d4 = (idx & 31) * 4;
        vreg[i] = *(const uint2*)(vrow + (size_t)(kt + t) * HD + d4);
      }
#pragma unroll
      for (int i = 0; i < 8; ++i) {
        int idx = tid + i * 256;
        int t = idx >> 5, d4 = (idx & 31) * 4;
        bf16 tmp[4];
        *(uint2*)tmp = vreg[i];
        Vt[(d4 + 0) * 64 + t] = tmp[0];
        Vt[(d4 + 1) * 64 + t] = tmp[1];
        Vt[(d4 + 2) * 64 + t] = tmp[2];
        Vt[(d4 + 3) * 64 + t] = tmp[3];
      }
    }
#ifdef HAVE_TDM
    if (wave == 0) __builtin_amdgcn_s_wait_tensorcnt(0);  // K tile kt landed
#endif
    __syncthreads();            // publish Ks[buf] + Vt to the workgroup
#ifdef HAVE_TDM
    if (wave == 0 && kt + 64 < ktend) {  // prefetch next K tile via TDM
      tdm_issue_2d((unsigned)(size_t)&Ks[buf ^ 1][0],
                   krow + (size_t)(kt + 64) * HD, HD, 64, HD);
    }
#endif

    // S = Q K^T  (16 x 64 per wave)
    v8f s[4] = {z8, z8, z8, z8};
#pragma unroll
    for (int kk = 0; kk < 4; ++kk) {
#pragma unroll
      for (int j = 0; j < 4; ++j) {
        v16bf bfrag = load_b_frag(Ks[buf], 128, j * 16, kk * 32);
        s[j] = wmma_bf16(qf[kk], bfrag, s[j]);
      }
    }

    // scale + causal mask
#pragma unroll
    for (int j = 0; j < 4; ++j) {
#pragma unroll
      for (int r = 0; r < 8; ++r) {
        int mg = q0 + wave * 16 + r + 8 * half;
        int ng = kt + j * 16 + (lane & 15);
        float sv = s[j][r] * scale;
        s[j][r] = (ng > mg) ? -1e30f : sv;
      }
    }

    // Online softmax per row (r,half); 16-lane butterfly reductions.
#pragma unroll
    for (int r = 0; r < 8; ++r) {
      float rm = fmaxf(fmaxf(s[0][r], s[1][r]), fmaxf(s[2][r], s[3][r]));
#pragma unroll
      for (int off = 8; off; off >>= 1) rm = fmaxf(rm, __shfl_xor(rm, off, 32));
      float mnew  = fmaxf(mrow[r], rm);
      float alpha = __expf(mrow[r] - mnew);
      mrow[r] = mnew;
      float rs = 0.f;
#pragma unroll
      for (int j = 0; j < 4; ++j) {
        float p = __expf(s[j][r] - mnew);
        s[j][r] = p;
        rs += p;
      }
#pragma unroll
      for (int off = 8; off; off >>= 1) rs += __shfl_xor(rs, off, 32);
      lrow[r] = lrow[r] * alpha + rs;
#pragma unroll
      for (int j = 0; j < 8; ++j) o[j][r] *= alpha;
    }

    // Stage P (bf16) through LDS to convert C-layout -> A-layout.
    bf16* pw = Ps + wave * (16 * 64);
#pragma unroll
    for (int j = 0; j < 4; ++j) {
#pragma unroll
      for (int r = 0; r < 8; ++r) {
        int m = r + 8 * half;
        int n = j * 16 + (lane & 15);
        pw[m * 64 + n] = (bf16)s[j][r];
      }
    }

    // O += P V  (V^T staged as [d][keypos], so B-frag lane = d)
#pragma unroll
    for (int kk2 = 0; kk2 < 2; ++kk2) {
      v16bf af = load_a_frag(pw, 64, 0, kk2 * 32);
#pragma unroll
      for (int j2 = 0; j2 < 8; ++j2) {
        v16bf bfrag = load_b_frag(Vt, 64, j2 * 16, kk2 * 32);
        o[j2] = wmma_bf16(af, bfrag, o[j2]);
      }
    }
    __syncthreads();            // done with Vt/Ks[buf] before restaging
    buf ^= 1;
  }

  // Epilogue: normalize and write O to [B][T][H*HD] bf16.
#pragma unroll
  for (int j2 = 0; j2 < 8; ++j2) {
#pragma unroll
    for (int r = 0; r < 8; ++r) {
      int t = q0 + wave * 16 + r + 8 * half;
      int d = j2 * 16 + (lane & 15);
      float val = o[j2][r] / lrow[r];
      ob[(size_t)(b * TT + t) * DD + h * HD + d] = (bf16)val;
    }
  }
}

// ---------------------------------------------------------------------------
// Kernel 5: out = O @ Wo^T + bo  (f32 output); double-buffered LDS pipeline.
// ---------------------------------------------------------------------------
__global__ __launch_bounds__(256) void gemm_out_kernel(
    const bf16* __restrict__ A, const float* __restrict__ w,
    const float* __restrict__ bias, float* __restrict__ out) {
  __shared__ bf16 As[2][128 * 32];
  __shared__ bf16 Bs[2][64 * 32];
  const int tid  = threadIdx.x;
  const int wave = tid >> 5;
  const int lane = tid & 31;
  const int half = lane >> 4;
  const int m0 = blockIdx.x * 128;
  const int n0 = blockIdx.y * 64;

  const v8f z8 = {0.f, 0.f, 0.f, 0.f, 0.f, 0.f, 0.f, 0.f};
  v8f acc[4] = {z8, z8, z8, z8};

  uint4 ra[2];
  float4 fb[2];
  auto load_tiles = [&](int k0) {
#pragma unroll
    for (int i = 0; i < 2; ++i) {       // A: straight bf16 copy, 8 elems/ld
      int e = (tid + i * 256) * 8;
      int r = e >> 5, c = e & 31;
      ra[i] = *(const uint4*)(A + (size_t)(m0 + r) * DD + k0 + c);
    }
#pragma unroll
    for (int i = 0; i < 2; ++i) {       // B: f32 -> bf16
      int e = (tid + i * 256) * 4;
      int r = e >> 5, c = e & 31;
      fb[i] = *(const float4*)(w + (size_t)(n0 + r) * DD + k0 + c);
    }
  };
  auto store_tiles = [&](int buf) {
#pragma unroll
    for (int i = 0; i < 2; ++i) {
      int e = (tid + i * 256) * 8;
      *(uint4*)(&As[buf][e]) = ra[i];
    }
#pragma unroll
    for (int i = 0; i < 2; ++i) {
      int e = (tid + i * 256) * 4;
      cvt4(&Bs[buf][e], fb[i]);
    }
  };

  load_tiles(0);
  store_tiles(0);
  int buf = 0;
  for (int k0 = 0; k0 < DD; k0 += 32) {
    __syncthreads();
    const bool more = (k0 + 32 < DD);
    if (more) load_tiles(k0 + 32);
    v16bf a = load_a_frag(As[buf], 32, wave * 16, 0);
#pragma unroll
    for (int j = 0; j < 4; ++j) {
      v16bf b = load_b_frag(Bs[buf], 32, j * 16, 0);
      acc[j] = wmma_bf16(a, b, acc[j]);
    }
    if (more) { buf ^= 1; store_tiles(buf); }
  }

#pragma unroll
  for (int j = 0; j < 4; ++j) {
#pragma unroll
    for (int r = 0; r < 8; ++r) {
      int m = m0 + wave * 16 + r + 8 * half;
      int n = n0 + j * 16 + (lane & 15);
      out[(size_t)m * DD + n] = acc[j][r] + bias[n];
    }
  }
}

// ---------------------------------------------------------------------------
// Launch
// ---------------------------------------------------------------------------
extern "C" void kernel_launch(void* const* d_in, const int* in_sizes, int n_in,
                              void* d_out, int out_size, void* d_ws,
                              size_t ws_size, hipStream_t stream) {
  const float* x    = (const float*)d_in[0];
  const float* Wqkv = (const float*)d_in[1];
  const float* bqkv = (const float*)d_in[2];
  const float* Wo   = (const float*)d_in[3];
  const float* bo   = (const float*)d_in[4];
  // d_in[5] = mask: causal, handled analytically.
  float* out = (float*)d_out;

  const size_t NQKV = (size_t)BB * HH * TT * HD;  // 8,388,608 elements
  char* ws = (char*)d_ws;
  size_t off = 0;
  bf16* qb = (bf16*)(ws + off); off += NQKV * sizeof(bf16);
  bf16* kb = (bf16*)(ws + off); off += NQKV * sizeof(bf16);
  bf16* vb = (bf16*)(ws + off); off += NQKV * sizeof(bf16);
  bf16* ob = (bf16*)(ws + off); off += (size_t)BT * DD * sizeof(bf16);
  float* ct = (float*)(ws + off); off += (size_t)TT * 64 * sizeof(float);
  float* st = (float*)(ws + off); off += (size_t)TT * 64 * sizeof(float);

  rope_table_kernel<<<TT, 64, 0, stream>>>(ct, st);
  gemm_qkv_kernel<<<dim3(BT / 128, D3 / 64), 256, 0, stream>>>(
      x, Wqkv, bqkv, qb, kb, vb);
  rope_apply_kernel<<<(BB * HH * TT * 64) / 256, 256, 0, stream>>>(
      qb, kb, ct, st);
  attn_kernel<<<dim3(TT / 128, BB * HH), 256, 0, stream>>>(qb, kb, vb, ob);
  gemm_out_kernel<<<dim3(BT / 128, DD / 64), 256, 0, stream>>>(ob, Wo, bo, out);
}
